// GATLayerEdgeSoftmax_3624952398656
// MI455X (gfx1250) — matrix operand order repaired and checked by reading, not compiled, not run on hardware
//
#include <hip/hip_runtime.h>
#include <hip/hip_bf16.h>
#include <math.h>

#define N_NODES 2048
#define N_EDGES 65536
#define D_IN    128
#define D_H     256      // 2*D_IN
#define D_OUT   128
#define EPS_F   1e-6f

typedef __attribute__((ext_vector_type(16))) __bf16 v16bf;
typedef __attribute__((ext_vector_type(8)))  float  v8f;
typedef __attribute__((ext_vector_type(4)))  unsigned int v4u;

// 16 bf16 (= one WMMA A/B fragment) viewed as two 16-byte chunks.
union Frag32 {
  v4u   q[2];
  v16bf v;
};

// Monotonic order-preserving uint key for float (for atomicMax-based global max)
__device__ __forceinline__ unsigned fkey(float f) {
  unsigned u = __float_as_uint(f);
  return (u & 0x80000000u) ? ~u : (u | 0x80000000u);
}
__device__ __forceinline__ float funkey(unsigned k) {
  unsigned u = (k & 0x80000000u) ? (k & 0x7fffffffu) : ~k;
  return __uint_as_float(u);
}

// ---------------- prep kernels ----------------

__global__ __launch_bounds__(256) void gat_init(float* acc, float* asum, unsigned* mk) {
  int i = blockIdx.x * 256 + threadIdx.x;           // grid covers N_NODES*D_OUT
  if (i < N_NODES * D_OUT) acc[i] = 0.f;
  if (i < N_NODES)         asum[i] = 0.f;
  if (i == 0)              *mk = 0u;
}

__global__ __launch_bounds__(256) void gat_cvt_x(const float* __restrict__ x,
                                                 __bf16* __restrict__ xb) {
  int i = blockIdx.x * 256 + threadIdx.x;           // N_NODES*D_IN
  xb[i] = (__bf16)x[i];
}

// fW is [256,128] row-major (K-major). Store bf16 transposed: fWbT[n][k], n in [0,128)
__global__ __launch_bounds__(256) void gat_cvt_fw(const float* __restrict__ fW,
                                                  __bf16* __restrict__ fWbT) {
  int i = blockIdx.x * 256 + threadIdx.x;           // D_H*D_OUT
  int k = i >> 7;                                   // 0..255
  int n = i & 127;                                  // 0..127
  fWbT[(size_t)n * D_H + k] = (__bf16)fW[i];
}

// ---------------- logits + global max ----------------

__global__ __launch_bounds__(256) void gat_logits(const float* __restrict__ x,
                                                  const int*   __restrict__ src,
                                                  const int*   __restrict__ tgt,
                                                  const float* __restrict__ wW,
                                                  const float* __restrict__ wb,
                                                  float* __restrict__ a,
                                                  unsigned* __restrict__ mk) {
  int e = blockIdx.x * 256 + threadIdx.x;           // E threads
  const float* xs = x + (size_t)src[e] * D_IN;
  const float* xt = x + (size_t)tgt[e] * D_IN;
  float s = wb[0];
  #pragma unroll 4
  for (int j = 0; j < D_IN; ++j) s = fmaf(xs[j], wW[j], s);
  #pragma unroll 4
  for (int j = 0; j < D_IN; ++j) s = fmaf(xt[j], wW[D_IN + j], s);
  a[e] = s;

  __shared__ unsigned red[256];
  red[threadIdx.x] = fkey(s);
  __syncthreads();
  for (int off = 128; off > 0; off >>= 1) {
    if (threadIdx.x < off) {
      unsigned o = red[threadIdx.x + off];
      if (o > red[threadIdx.x]) red[threadIdx.x] = o;
    }
    __syncthreads();
  }
  if (threadIdx.x == 0) atomicMax(mk, red[0]);
}

// ---------------- exp + per-target denominator ----------------

__global__ __launch_bounds__(256) void gat_expsum(const float* __restrict__ a,
                                                  const unsigned* __restrict__ mk,
                                                  const int* __restrict__ tgt,
                                                  float* __restrict__ aexp,
                                                  float* __restrict__ asum) {
  int e = blockIdx.x * 256 + threadIdx.x;
  float amax = funkey(*mk);
  float v = expf(a[e] - amax);
  aexp[e] = v;
  atomicAdd(asum + tgt[e], v);
}

// ---------------- main fused WMMA kernel ----------------
// One wave -> one 16-edge x 32-col output tile of y=relu(h@fW+fb),
// scaled by a_exp and scatter-added into acc[tgt].
__global__ __launch_bounds__(256) void gat_edge_wmma(const __bf16* __restrict__ xb,
                                                     const __bf16* __restrict__ fWbT,
                                                     const float*  __restrict__ fb,
                                                     const float*  __restrict__ aexp,
                                                     const int*    __restrict__ src,
                                                     const int*    __restrict__ tgt,
                                                     float* __restrict__ acc) {
  const int wid   = (blockIdx.x * blockDim.x + threadIdx.x) >> 5;
  const int lane  = threadIdx.x & 31;
  const int etile = wid >> 2;                 // 4096 edge tiles of 16
  const int cpair = wid & 3;                  // 4 pairs of 16-col tiles
  const int e0 = etile << 4;
  const int n0 = cpair << 5;

  const int mrow  = lane & 15;
  const int hi    = lane >> 4;                // half-wave select
  const int hioff = hi << 3;                  // A: K sub-offset {0,8}
  const int kboff = hi << 3;                  // B: K sub-offset {0,8}

  const int eA   = e0 + mrow;                 // edge for this lane's A row
  const int nsrc = src[eA];
  const int ntgt = tgt[eA];

  const int ncol0 = n0 + mrow;                // output columns held by this lane
  const int ncol1 = ncol0 + 16;

  v8f c0 = {};
  v8f c1 = {};

  #pragma unroll
  for (int ks = 0; ks < 8; ++ks) {
    const int k0 = ks << 5;                   // K block start in h (0..224)
    // h = [x[src] | x[tgt]]: first 128 K's come from src row, rest from tgt row
    const int node = (k0 < D_IN) ? nsrc : ntgt;
    const int kk   = k0 & (D_IN - 1);

    // A fragment: lane half 0 holds K = {k0+0..7, k0+16..23}; half 1 holds {+8..15, +24..31}
    const __bf16* pa = xb + (size_t)node * D_IN + kk + hioff;
    Frag32 af;
    af.q[0] = *(const v4u*)(pa);
    af.q[1] = *(const v4u*)(pa + 16);

    // B fragments: lane half 0 holds K = {k0+0..7, k0+16..23} of its column; half 1 holds {+8..15, +24..31}
    const __bf16* pb0 = fWbT + (size_t)ncol0 * D_H + k0 + kboff;
    const __bf16* pb1 = fWbT + (size_t)ncol1 * D_H + k0 + kboff;
    Frag32 bf0, bf1;
    bf0.q[0] = *(const v4u*)(pb0);
    bf0.q[1] = *(const v4u*)(pb0 + 16);
    bf1.q[0] = *(const v4u*)(pb1);
    bf1.q[1] = *(const v4u*)(pb1 + 16);

    c0 = __builtin_amdgcn_wmma_f32_16x16x32_bf16(false, af.v, false, bf0.v,
                                                 (short)0, c0, false, false);
    c1 = __builtin_amdgcn_wmma_f32_16x16x32_bf16(false, af.v, false, bf1.v,
                                                 (short)0, c1, false, false);
  }

  // C/D layout: VGPR r, lanes 0-15 -> M=r; lanes 16-31 -> M=8+r; N = lane%16
  const float bias0 = fb[ncol0];
  const float bias1 = fb[ncol1];
  #pragma unroll
  for (int r = 0; r < 8; ++r) {
    const int m = r + (hi << 3);
    const int e = e0 + m;
    const float w = aexp[e];
    const int   t = tgt[e];
    float v0 = fmaxf(c0[r] + bias0, 0.f) * w;
    float v1 = fmaxf(c1[r] + bias1, 0.f) * w;
    atomicAdd(acc + (size_t)t * D_OUT + ncol0, v0);
    atomicAdd(acc + (size_t)t * D_OUT + ncol1, v1);
  }
}

// ---------------- finalize ----------------

__global__ __launch_bounds__(256) void gat_final(const float* __restrict__ acc,
                                                 const float* __restrict__ asum,
                                                 float* __restrict__ out) {
  int i = blockIdx.x * 256 + threadIdx.x;     // N_NODES*D_OUT
  int n = i >> 7;
  out[i] = acc[i] / (asum[n] + EPS_F);
}

// ---------------- host launcher ----------------

extern "C" void kernel_launch(void* const* d_in, const int* in_sizes, int n_in,
                              void* d_out, int out_size, void* d_ws, size_t ws_size,
                              hipStream_t stream) {
  // setup_inputs order: x, adj, src, tgt, Msrc, Mtgt, fW, fb, wW, wb
  const float* x   = (const float*)d_in[0];
  const int*   src = (const int*)d_in[2];
  const int*   tgt = (const int*)d_in[3];
  const float* fW  = (const float*)d_in[6];
  const float* fb  = (const float*)d_in[7];
  const float* wW  = (const float*)d_in[8];
  const float* wb  = (const float*)d_in[9];
  float* out = (float*)d_out;

  char* ws = (char*)d_ws;
  __bf16*   xb   = (__bf16*)(ws);                 // 2048*128*2  = 512 KB
  __bf16*   fWbT = (__bf16*)(ws + 524288);        // 128*256*2   =  64 KB
  float*    a    = (float*) (ws + 589824);        // 65536*4     = 256 KB
  float*    aexp = (float*) (ws + 851968);        // 65536*4     = 256 KB
  float*    asum = (float*) (ws + 1114112);       // 2048*4      =   8 KB
  unsigned* mk   = (unsigned*)(ws + 1122304);     // 4 B (padded to 512)
  float*    acc  = (float*) (ws + 1122816);       // 2048*128*4  =   1 MB

  gat_init     <<<1024, 256, 0, stream>>>(acc, asum, mk);
  gat_cvt_x    <<<1024, 256, 0, stream>>>(x, xb);
  gat_cvt_fw   <<< 128, 256, 0, stream>>>(fW, fWbT);
  gat_logits   <<< 256, 256, 0, stream>>>(x, src, tgt, wW, wb, a, mk);
  gat_expsum   <<< 256, 256, 0, stream>>>(a, mk, tgt, aexp, asum);
  // 4096 edge tiles * 4 col-pairs = 16384 waves = 2048 blocks * 8 waves
  gat_edge_wmma<<<2048, 256, 0, stream>>>(xb, fWbT, fb, aexp, src, tgt, acc);
  gat_final    <<<1024, 256, 0, stream>>>(acc, asum, out);
}
